// MultiheadAttentionC_9929964388664
// MI455X (gfx1250) — matrix operand order repaired and checked by reading, not compiled, not run on hardware
//
#include <hip/hip_runtime.h>
#include <cstdint>
#include <cstddef>

// ---------------------------------------------------------------------------
// MHA forward for MI455X (gfx1250, wave32).  bf16 WMMA (v_wmma_f32_16x16x32_bf16)
// for all GEMMs, f32 accumulate, f32 softmax.  p (attn probs) is a required
// output -> written straight into d_out's p region and normalized in place.
// Tile staging uses GLOBAL_LOAD_ASYNC_TO_LDS_B128 (ASYNCcnt) when available.
// ---------------------------------------------------------------------------

typedef __bf16        v16bf __attribute__((ext_vector_type(16)));
typedef float         v8f   __attribute__((ext_vector_type(8)));
typedef unsigned int  v8u   __attribute__((ext_vector_type(8)));
typedef int           v4i   __attribute__((ext_vector_type(4)));
typedef unsigned short u16;

static constexpr int T_  = 2048;
static constexpr int B_  = 2;
static constexpr int E_  = 1024;
static constexpr int H_  = 16;
static constexpr int HD_ = 64;      // head dim
static constexpr int BH_ = B_ * H_; // 32
static constexpr int MROW = T_ * B_;   // 4096 rows for projections
static constexpr int N3E  = 3 * E_;    // 3072

#define DEVINL static __device__ __forceinline__

// ---- async global -> LDS staging (CDNA5), with synchronous fallback -------
#if defined(__has_builtin)
#  if __has_builtin(__builtin_amdgcn_global_load_async_to_lds_b128)
#    define HAVE_ASYNC_LDS 1
#  endif
#endif
#ifndef HAVE_ASYNC_LDS
#  define HAVE_ASYNC_LDS 0
#endif

DEVINL void stage16B(const u16* gp, u16* lp) {
#if HAVE_ASYNC_LDS
  // 16 bytes per lane, global -> LDS, tracked by ASYNCcnt
  __builtin_amdgcn_global_load_async_to_lds_b128((v4i*)gp, (v4i*)lp, 0, 0);
#else
  *(uint4*)lp = *(const uint4*)gp;
#endif
}

DEVINL void stage_join() {
#if HAVE_ASYNC_LDS
#  if __has_builtin(__builtin_amdgcn_s_wait_asynccnt)
  __builtin_amdgcn_s_wait_asynccnt(0);
#  else
  asm volatile("s_wait_asynccnt 0x0" ::: "memory");
#  endif
#endif
}

DEVINL u16 f2bf(float f) {
  unsigned int u = __builtin_bit_cast(unsigned int, f);
  unsigned int r = u + 0x7FFFu + ((u >> 16) & 1u);   // round-to-nearest-even
  return (u16)(r >> 16);
}

DEVINL v16bf pack_frag(uint4 lo, uint4 hi) {
  v8u u = {lo.x, lo.y, lo.z, lo.w, hi.x, hi.y, hi.z, hi.w};
  return __builtin_bit_cast(v16bf, u);
}

DEVINL v8f wmma_bf16(v16bf a, v16bf b, v8f c) {
  // 8 args: (neg_a, A, neg_b, B, c_mod, C, reuse_a, reuse_b)
  return __builtin_amdgcn_wmma_f32_16x16x32_bf16(false, a, false, b,
                                                 (short)0, c, false, false);
}

struct alignas(8) us4 { u16 x, y, z, w; };

// ------------------------------- f32 -> bf16 -------------------------------
__global__ __launch_bounds__(256) void cvt_f32_bf16(const float* __restrict__ src,
                                                    u16* __restrict__ dst, int n4) {
  int i = (blockIdx.x * 256 + threadIdx.x);
  if (i < n4) {
    float4 v = reinterpret_cast<const float4*>(src)[i];
    us4 o = { f2bf(v.x), f2bf(v.y), f2bf(v.z), f2bf(v.w) };
    reinterpret_cast<us4*>(dst)[i] = o;
  }
}

// ------------------------- QKV projection GEMM -----------------------------
// C[m,n] = sum_k g[m,k] * W_in[n,k] + b_in[n]   (M=4096, N=3072, K=1024)
// Epilogue scatters q (scaled), k into [BH,T,64]; v transposed into [BH,64,T].
__global__ __launch_bounds__(256) void gemm_qkv(const u16* __restrict__ gbf,
                                                const u16* __restrict__ wibf,
                                                const float* __restrict__ b_in,
                                                u16* __restrict__ qws,
                                                u16* __restrict__ kws,
                                                u16* __restrict__ vtws) {
  constexpr int LDA = 40;                 // 32 + pad
  __shared__ u16 As[128 * LDA];
  __shared__ u16 Bs[128 * LDA];

  const int tid  = threadIdx.x;
  const int lane = tid & 31;
  const int wid  = tid >> 5;
  const int waveM = wid >> 2;             // 0..1 (64 rows each)
  const int waveN = wid & 3;              // 0..3 (32 cols each)
  const int g  = lane >> 4;               // lane-hi group
  const int ln = lane & 15;
  const int M0 = blockIdx.y * 128;
  const int N0 = blockIdx.x * 128;

  const int ldr = tid >> 1;               // 0..127
  const int ldc = (tid & 1) * 16;         // 0 / 16

  v8f c[4][2] = {};

  for (int k0 = 0; k0 < E_; k0 += 32) {
    const u16* sa = gbf  + (size_t)(M0 + ldr) * E_ + k0 + ldc;
    const u16* sb = wibf + (size_t)(N0 + ldr) * E_ + k0 + ldc;
    stage16B(sa,     &As[ldr * LDA + ldc]);
    stage16B(sa + 8, &As[ldr * LDA + ldc + 8]);
    stage16B(sb,     &Bs[ldr * LDA + ldc]);
    stage16B(sb + 8, &Bs[ldr * LDA + ldc + 8]);
    stage_join();
    __syncthreads();

    v16bf af[4], bfr[2];
#pragma unroll
    for (int mt = 0; mt < 4; ++mt) {
      int m = waveM * 64 + mt * 16 + ln;
      af[mt] = pack_frag(*(const uint4*)&As[m * LDA + g * 8],
                         *(const uint4*)&As[m * LDA + 16 + g * 8]);
    }
#pragma unroll
    for (int nt = 0; nt < 2; ++nt) {
      int n = waveN * 32 + nt * 16 + ln;
      bfr[nt] = pack_frag(*(const uint4*)&Bs[n * LDA + g * 8],
                          *(const uint4*)&Bs[n * LDA + 16 + g * 8]);
    }
#pragma unroll
    for (int mt = 0; mt < 4; ++mt)
#pragma unroll
      for (int nt = 0; nt < 2; ++nt)
        c[mt][nt] = wmma_bf16(af[mt], bfr[nt], c[mt][nt]);
    __syncthreads();
  }

  // epilogue: bias + scatter to head-major q/k and transposed v
#pragma unroll
  for (int mt = 0; mt < 4; ++mt) {
#pragma unroll
    for (int nt = 0; nt < 2; ++nt) {
      const int ntile = N0 + waveN * 32 + nt * 16;        // tile-uniform
      const int sect  = ntile >> 10;                      // 0=q 1=k 2=v
      const int ncol  = ntile + ln;
      const float bias = b_in[ncol];
      const int e = ncol & 1023;
      const int h = e >> 6, d = e & 63;
      const int mbase = M0 + waveM * 64 + mt * 16 + 8 * g;
      if (sect == 0) {
#pragma unroll
        for (int r = 0; r < 8; ++r) {
          const int m = mbase + r;
          const int bh = (m & 1) * H_ + h;
          qws[((size_t)bh * T_ + (m >> 1)) * HD_ + d] =
              f2bf((c[mt][nt][r] + bias) * 0.125f);       // hd^-0.5, hd=64
        }
      } else if (sect == 1) {
#pragma unroll
        for (int r = 0; r < 8; ++r) {
          const int m = mbase + r;
          const int bh = (m & 1) * H_ + h;
          kws[((size_t)bh * T_ + (m >> 1)) * HD_ + d] = f2bf(c[mt][nt][r] + bias);
        }
      } else {
#pragma unroll
        for (int r = 0; r < 8; ++r) {
          const int m = mbase + r;
          const int bh = (m & 1) * H_ + h;
          vtws[(size_t)bh * HD_ * T_ + (size_t)d * T_ + (m >> 1)] =
              f2bf(c[mt][nt][r] + bias);
        }
      }
    }
  }
}

// --------------------------- scores = q @ k^T ------------------------------
// per head: M=T, N=T, K=64.  Raw scores -> p region (f32), softmax later.
__global__ __launch_bounds__(256) void gemm_scores(const u16* __restrict__ qws,
                                                   const u16* __restrict__ kws,
                                                   float* __restrict__ p) {
  constexpr int LDA = 72;                 // 64 + pad
  __shared__ u16 As[128 * LDA];
  __shared__ u16 Bs[128 * LDA];

  const int tid  = threadIdx.x;
  const int lane = tid & 31;
  const int wid  = tid >> 5;
  const int waveM = wid >> 2, waveN = wid & 3;
  const int g = lane >> 4, ln = lane & 15;
  const int bh = blockIdx.z;
  const int M0 = blockIdx.y * 128;
  const int N0 = blockIdx.x * 128;

  const int ldr = tid >> 1;
  const int ldh = (tid & 1) * 32;

  {
    const u16* sq = qws + ((size_t)bh * T_ + M0 + ldr) * HD_ + ldh;
    const u16* sk = kws + ((size_t)bh * T_ + N0 + ldr) * HD_ + ldh;
#pragma unroll
    for (int i = 0; i < 4; ++i) {
      stage16B(sq + i * 8, &As[ldr * LDA + ldh + i * 8]);
      stage16B(sk + i * 8, &Bs[ldr * LDA + ldh + i * 8]);
    }
  }
  stage_join();
  __syncthreads();

  v8f c[4][2] = {};
#pragma unroll
  for (int ks = 0; ks < 2; ++ks) {
    v16bf af[4], bfr[2];
#pragma unroll
    for (int mt = 0; mt < 4; ++mt) {
      int m = waveM * 64 + mt * 16 + ln;
      af[mt] = pack_frag(*(const uint4*)&As[m * LDA + ks * 32 + g * 8],
                         *(const uint4*)&As[m * LDA + ks * 32 + 16 + g * 8]);
    }
#pragma unroll
    for (int nt = 0; nt < 2; ++nt) {
      int n = waveN * 32 + nt * 16 + ln;
      bfr[nt] = pack_frag(*(const uint4*)&Bs[n * LDA + ks * 32 + g * 8],
                          *(const uint4*)&Bs[n * LDA + ks * 32 + 16 + g * 8]);
    }
#pragma unroll
    for (int mt = 0; mt < 4; ++mt)
#pragma unroll
      for (int nt = 0; nt < 2; ++nt)
        c[mt][nt] = wmma_bf16(af[mt], bfr[nt], c[mt][nt]);
  }

#pragma unroll
  for (int mt = 0; mt < 4; ++mt)
#pragma unroll
    for (int nt = 0; nt < 2; ++nt) {
      const int n = N0 + waveN * 32 + nt * 16 + ln;
      const int mbase = M0 + waveM * 64 + mt * 16 + 8 * g;
#pragma unroll
      for (int r = 0; r < 8; ++r) {
        size_t idx = ((size_t)bh * T_ + (mbase + r)) * T_ + n;
        p[idx] = c[mt][nt][r];
      }
    }
}

// --------------------------- row softmax (in place) ------------------------
__global__ __launch_bounds__(256) void softmax_rows(float* __restrict__ p) {
  __shared__ float red[256];
  const int tid = threadIdx.x;
  float* row = p + (size_t)blockIdx.x * T_;

  float4 x0 = *(const float4*)(row + tid * 4);
  float4 x1 = *(const float4*)(row + 1024 + tid * 4);

  float lm = fmaxf(fmaxf(fmaxf(x0.x, x0.y), fmaxf(x0.z, x0.w)),
                   fmaxf(fmaxf(x1.x, x1.y), fmaxf(x1.z, x1.w)));
  red[tid] = lm;
  __syncthreads();
#pragma unroll
  for (int s = 128; s > 0; s >>= 1) {
    if (tid < s) red[tid] = fmaxf(red[tid], red[tid + s]);
    __syncthreads();
  }
  const float mx = red[0];
  __syncthreads();

  x0.x = __expf(x0.x - mx); x0.y = __expf(x0.y - mx);
  x0.z = __expf(x0.z - mx); x0.w = __expf(x0.w - mx);
  x1.x = __expf(x1.x - mx); x1.y = __expf(x1.y - mx);
  x1.z = __expf(x1.z - mx); x1.w = __expf(x1.w - mx);

  red[tid] = (x0.x + x0.y + x0.z + x0.w) + (x1.x + x1.y + x1.z + x1.w);
  __syncthreads();
#pragma unroll
  for (int s = 128; s > 0; s >>= 1) {
    if (tid < s) red[tid] += red[tid + s];
    __syncthreads();
  }
  const float inv = 1.0f / red[0];

  x0.x *= inv; x0.y *= inv; x0.z *= inv; x0.w *= inv;
  x1.x *= inv; x1.y *= inv; x1.z *= inv; x1.w *= inv;
  *(float4*)(row + tid * 4)        = x0;
  *(float4*)(row + 1024 + tid * 4) = x1;
}

// ------------------------------ attn = p @ v -------------------------------
// per head: M=T, N=64, K=T.  A staged f32->bf16 via VGPRs; B async from vT.
// Writes attn in [T, B, E] layout for the output projection.
__global__ __launch_bounds__(256) void gemm_pv(const float* __restrict__ p,
                                               const u16* __restrict__ vtws,
                                               u16* __restrict__ attnws) {
  constexpr int LDA = 72;
  __shared__ u16 As[128 * LDA];
  __shared__ u16 Bs[64 * LDA];

  const int tid  = threadIdx.x;
  const int lane = tid & 31;
  const int wid  = tid >> 5;                 // wave row 0..7 (16 M each)
  const int g = lane >> 4, ln = lane & 15;
  const int bh = blockIdx.z;
  const int M0 = blockIdx.x * 128;

  v8f c[4] = {};

  for (int k0 = 0; k0 < T_; k0 += 64) {
    // B: vT rows, contiguous along K (async -> LDS)
    {
      const int ldr = tid >> 2;               // 0..63
      const int ldq = (tid & 3) * 16;
      const u16* sv = vtws + (size_t)bh * HD_ * T_ + (size_t)ldr * T_ + k0 + ldq;
      stage16B(sv,     &Bs[ldr * LDA + ldq]);
      stage16B(sv + 8, &Bs[ldr * LDA + ldq + 8]);
    }
    // A: 128 x 64 f32 -> bf16 (through VGPRs: needs conversion)
    {
      const int ldr = tid >> 1;
      const int ldh = (tid & 1) * 32;
      const float* sp = p + ((size_t)bh * T_ + M0 + ldr) * T_ + k0 + ldh;
      __builtin_prefetch(sp + 64, 0, 1);
#pragma unroll
      for (int i = 0; i < 8; ++i) {
        float4 v = *(const float4*)(sp + i * 4);
        us4 o = { f2bf(v.x), f2bf(v.y), f2bf(v.z), f2bf(v.w) };
        *(us4*)&As[ldr * LDA + ldh + i * 4] = o;
      }
    }
    stage_join();
    __syncthreads();

#pragma unroll
    for (int ks = 0; ks < 2; ++ks) {
      const int m = wid * 16 + ln;
      v16bf af = pack_frag(*(const uint4*)&As[m * LDA + ks * 32 + g * 8],
                           *(const uint4*)&As[m * LDA + ks * 32 + 16 + g * 8]);
#pragma unroll
      for (int nt = 0; nt < 4; ++nt) {
        const int n = nt * 16 + ln;
        v16bf bfr = pack_frag(*(const uint4*)&Bs[n * LDA + ks * 32 + g * 8],
                              *(const uint4*)&Bs[n * LDA + ks * 32 + 16 + g * 8]);
        c[nt] = wmma_bf16(af, bfr, c[nt]);
      }
    }
    __syncthreads();
  }

  const int b = bh >> 4, h = bh & 15;
#pragma unroll
  for (int nt = 0; nt < 4; ++nt) {
    const int d = nt * 16 + ln;
    const int mbase = M0 + wid * 16 + 8 * g;
#pragma unroll
    for (int r = 0; r < 8; ++r) {
      const int t = mbase + r;
      attnws[(size_t)t * (B_ * E_) + b * E_ + h * HD_ + d] = f2bf(c[nt][r]);
    }
  }
}

// --------------------------- output projection -----------------------------
// out[m,n] = sum_k attn[m,k] * W_out[n,k] + b_out[n]  (M=4096, N=1024, K=1024)
__global__ __launch_bounds__(256) void gemm_out(const u16* __restrict__ abf,
                                                const u16* __restrict__ wobf,
                                                const float* __restrict__ b_out,
                                                float* __restrict__ out) {
  constexpr int LDA = 40;
  __shared__ u16 As[128 * LDA];
  __shared__ u16 Bs[128 * LDA];

  const int tid  = threadIdx.x;
  const int lane = tid & 31;
  const int wid  = tid >> 5;
  const int waveM = wid >> 2, waveN = wid & 3;
  const int g = lane >> 4, ln = lane & 15;
  const int M0 = blockIdx.y * 128;
  const int N0 = blockIdx.x * 128;

  const int ldr = tid >> 1;
  const int ldc = (tid & 1) * 16;

  v8f c[4][2] = {};

  for (int k0 = 0; k0 < E_; k0 += 32) {
    const u16* sa = abf  + (size_t)(M0 + ldr) * E_ + k0 + ldc;
    const u16* sb = wobf + (size_t)(N0 + ldr) * E_ + k0 + ldc;
    stage16B(sa,     &As[ldr * LDA + ldc]);
    stage16B(sa + 8, &As[ldr * LDA + ldc + 8]);
    stage16B(sb,     &Bs[ldr * LDA + ldc]);
    stage16B(sb + 8, &Bs[ldr * LDA + ldc + 8]);
    stage_join();
    __syncthreads();

    v16bf af[4], bfr[2];
#pragma unroll
    for (int mt = 0; mt < 4; ++mt) {
      int m = waveM * 64 + mt * 16 + ln;
      af[mt] = pack_frag(*(const uint4*)&As[m * LDA + g * 8],
                         *(const uint4*)&As[m * LDA + 16 + g * 8]);
    }
#pragma unroll
    for (int nt = 0; nt < 2; ++nt) {
      int n = waveN * 32 + nt * 16 + ln;
      bfr[nt] = pack_frag(*(const uint4*)&Bs[n * LDA + g * 8],
                          *(const uint4*)&Bs[n * LDA + 16 + g * 8]);
    }
#pragma unroll
    for (int mt = 0; mt < 4; ++mt)
#pragma unroll
      for (int nt = 0; nt < 2; ++nt)
        c[mt][nt] = wmma_bf16(af[mt], bfr[nt], c[mt][nt]);
    __syncthreads();
  }

#pragma unroll
  for (int mt = 0; mt < 4; ++mt)
#pragma unroll
    for (int nt = 0; nt < 2; ++nt) {
      const int n = N0 + waveN * 32 + nt * 16 + ln;
      const float bias = b_out[n];
      const int mbase = M0 + waveM * 64 + mt * 16 + 8 * g;
#pragma unroll
      for (int r = 0; r < 8; ++r)
        out[(size_t)(mbase + r) * E_ + n] = c[mt][nt][r] + bias;
    }
}

// ---------------------------------------------------------------------------
extern "C" void kernel_launch(void* const* d_in, const int* in_sizes, int n_in,
                              void* d_out, int out_size, void* d_ws, size_t ws_size,
                              hipStream_t stream) {
  const float* g     = (const float*)d_in[0];   // [T,B,E]
  const float* W_in  = (const float*)d_in[1];   // [3E,E]
  const float* b_in  = (const float*)d_in[2];   // [3E]
  const float* W_out = (const float*)d_in[3];   // [E,E]
  const float* b_out = (const float*)d_in[4];   // [E]

  float* out = (float*)d_out;                               // [T,B,E]
  float* p   = (float*)d_out + (size_t)MROW * E_;           // [BH,T,T]

  // workspace layout (bf16 / u16 elements), total 48 MiB
  u16* gbf  = (u16*)d_ws;
  u16* wibf = gbf  + (size_t)MROW * E_;        // 4,194,304
  u16* wobf = wibf + (size_t)N3E  * E_;        // 3,145,728
  u16* qws  = wobf + (size_t)E_   * E_;        // 1,048,576
  u16* kws  = qws  + (size_t)BH_ * T_ * HD_;   // 4,194,304
  u16* vtws = kws  + (size_t)BH_ * T_ * HD_;
  u16* aws  = vtws + (size_t)BH_ * T_ * HD_;

  // 1) f32 -> bf16 conversions
  cvt_f32_bf16<<<(MROW * E_ / 4 + 255) / 256, 256, 0, stream>>>(g, gbf, MROW * E_ / 4);
  cvt_f32_bf16<<<(N3E * E_ / 4 + 255) / 256, 256, 0, stream>>>(W_in, wibf, N3E * E_ / 4);
  cvt_f32_bf16<<<(E_ * E_ / 4 + 255) / 256, 256, 0, stream>>>(W_out, wobf, E_ * E_ / 4);

  // 2) qkv = g @ W_in^T + b_in ; split/scale/scatter to head-major layouts
  gemm_qkv<<<dim3(N3E / 128, MROW / 128), 256, 0, stream>>>(gbf, wibf, b_in,
                                                            qws, kws, vtws);
  // 3) raw scores = q @ k^T per head (scale folded into q)
  gemm_scores<<<dim3(T_ / 128, T_ / 128, BH_), 256, 0, stream>>>(qws, kws, p);
  // 4) softmax rows in place -> p is the final probs output
  softmax_rows<<<dim3(BH_ * T_), 256, 0, stream>>>(p);
  // 5) attn = p @ v  -> [T,B,E] bf16
  gemm_pv<<<dim3(T_ / 128, 1, BH_), 256, 0, stream>>>(p, vtws, aws);
  // 6) out = attn @ W_out^T + b_out
  gemm_out<<<dim3(E_ / 128, MROW / 128), 256, 0, stream>>>(aws, wobf, b_out, out);
}
